// MultiHeadAttention_84447646974458
// MI455X (gfx1250) — matrix operand order repaired and checked by reading, not compiled, not run on hardware
//
#include <hip/hip_runtime.h>
#include <hip/hip_bf16.h>
#include <math.h>

// ---------------------------------------------------------------------------
// MHA forward for MI455X (gfx1250, wave32):
//   WMMA f32_16x16x32_f16 for all matmuls, TDM (tensor_load_to_lds) for all
//   global->LDS tile staging, double-buffered to overlap DMA with WMMA.
// B=2, S=2048, H=2048, NH=16, DH=128
// ---------------------------------------------------------------------------

typedef __attribute__((ext_vector_type(16))) _Float16 v16h;
typedef __attribute__((ext_vector_type(8)))  float    v8f;
typedef unsigned int u32x4 __attribute__((ext_vector_type(4)));
typedef int          i32x4 __attribute__((ext_vector_type(4)));
typedef int          i32x8 __attribute__((ext_vector_type(8)));

#define HID   2048
#define NHEAD 16
#define DH    128
#define BB    2
#define SS    2048
#define MROWS (BB * SS)     // 4096
#define NQKV  (3 * HID)     // 6144

#if __has_builtin(__builtin_amdgcn_tensor_load_to_lds) && \
    __has_builtin(__builtin_amdgcn_s_wait_tensorcnt)
#define HAVE_TDM 1
#else
#define HAVE_TDM 0
#endif

// ---------------- WMMA fragment helpers (contiguous k = 16h + s, same for A and B) ----------

// A fragment: 16x32 f16 (M x K), row-major src, leading dim ld.
// Lanes 0-15: row M=lane, halves 0..15=K0..15.
// Lanes 16-31: row M=lane-16, halves 0..15=K16..31.
__device__ __forceinline__ v16h load_frag_a(const _Float16* src, int ld) {
  const int lane = threadIdx.x & 31;
  const int m  = lane & 15;
  const int kb = (lane >> 4) << 4;
  v16h a;
#pragma unroll
  for (int i = 0; i < 16; ++i) a[i] = src[m * ld + kb + i];
  return a;
}

// B fragment: 32x16 (K x N) from N-major storage src[n*ld + k].
// Lane = column N; lanes 0-15 hold K0..15, lanes 16-31 hold K16..31.
__device__ __forceinline__ v16h load_frag_b_nmajor(const _Float16* src, int ld) {
  const int lane = threadIdx.x & 31;
  const int n  = lane & 15;
  const int kb = (lane >> 4) << 4;
  v16h b;
#pragma unroll
  for (int i = 0; i < 16; ++i) b[i] = src[n * ld + kb + i];
  return b;
}

// B fragment: 32x16 (K x N) from K-major storage src[k*ld + n].
__device__ __forceinline__ v16h load_frag_b_kmajor(const _Float16* src, int ld) {
  const int lane = threadIdx.x & 31;
  const int n  = lane & 15;
  const int kb = (lane >> 4) << 4;
  v16h b;
#pragma unroll
  for (int i = 0; i < 16; ++i) b[i] = src[(kb + i) * ld + n];
  return b;
}

__device__ __forceinline__ v8f wmma_f16(v16h a, v16h b, v8f c) {
  return __builtin_amdgcn_wmma_f32_16x16x32_f16(
      false, a, false, b, (short)0, c, false, false);
}

// ---------------- TDM 2-D tile load (D# per CDNA5 ISA) ---

#if HAVE_TDM
__device__ __forceinline__ unsigned int lds_off(const void* p) {
  return (unsigned int)(unsigned long long)
      (__attribute__((address_space(3))) const void*)p;
}

// Load a tile_d1 x tile_d0 tile of 2-byte elements whose top-left corner is
// gptr, row pitch stride0 (elements), into LDS at lds_byte (rows packed
// contiguously: tile_d0 elements per row).
__device__ __forceinline__ void tdm_load_2d(const _Float16* gptr,
                                            unsigned int lds_byte,
                                            unsigned int tensor_d0,
                                            unsigned int tensor_d1,
                                            unsigned int tile_d0,
                                            unsigned int tile_d1,
                                            unsigned int stride0) {
  const unsigned long long ga = (unsigned long long)gptr;
  u32x4 g0;
  g0.x = 1u;                                   // count=1, user mode, no gather
  g0.y = lds_byte;                             // D#[63:32]  lds_addr
  g0.z = (unsigned int)(ga & 0xFFFFFFFFull);   // D#[95:64]  global_addr lo
  g0.w = (unsigned int)((ga >> 32) & 0x1FFFFFFull) | 0x80000000u; // hi | type=2
  i32x8 g1;
  g1[0] = (int)(1u << 16);                     // data_size=1 (2 bytes)
  g1[1] = (int)((tensor_d0 & 0xFFFFu) << 16);                 // dim0[15:0]
  g1[2] = (int)(((tensor_d0 >> 16) & 0xFFFFu) |
                ((tensor_d1 & 0xFFFFu) << 16));               // dim0 hi|dim1 lo
  g1[3] = (int)(((tensor_d1 >> 16) & 0xFFFFu) | (tile_d0 << 16));
  g1[4] = (int)(tile_d1 & 0xFFFFu);            // tile_dim1 (tile_dim2=0)
  g1[5] = (int)stride0;                        // dim0_stride[31:0]
  g1[6] = 0;                                   // stride hi / dim1_stride lo
  g1[7] = 0;
  i32x4 gz4 = {0, 0, 0, 0};
  i32x8 gz8 = {0, 0, 0, 0, 0, 0, 0, 0};
  // 6-arg form (this toolchain): groups 0..3 + extra group + cpol.
  __builtin_amdgcn_tensor_load_to_lds(g0, g1, gz4, gz4, gz8, 0);
}
#else
// Fallback: vectorized b128 copy, rows*cols halves, cpr = cols/8 chunks/row.
__device__ __forceinline__ void stage_tile(const _Float16* g, int ldg,
                                           _Float16* lds, int nchunks,
                                           int cpr) {
  for (int idx = threadIdx.x; idx < nchunks; idx += 256) {
    int r = idx / cpr, c = idx % cpr;
    ((uint4*)lds)[idx] = *(const uint4*)(g + (size_t)r * ldg + c * 8);
  }
}
#endif

// ---------------- Kernel 1: fp32 -> f16 convert (vectorized) ----------------

__global__ void cvt_f32_f16(const float* __restrict__ in,
                            _Float16* __restrict__ out, int n4) {
  int i = blockIdx.x * blockDim.x + threadIdx.x;
  if (i < n4) {
    float4 v = ((const float4*)in)[i];
    union { _Float16 h[4]; uint2 u; } t;
    t.h[0] = (_Float16)v.x; t.h[1] = (_Float16)v.y;
    t.h[2] = (_Float16)v.z; t.h[3] = (_Float16)v.w;
    ((uint2*)out)[i] = t.u;
  }
}

// ---------------- Kernel 2: fused QKV projection (NT GEMM) ------------------
// C[m,o] = sum_h X[m,h] * W[o,h]; M=4096, N=6144, K=2048.
// Block tile 64x128, BK=32, TDM double-buffered staging; epilogue scatters
// into head-major Q/K/V [b][nh][s][dh] (f16).
__global__ __launch_bounds__(256) void qkv_gemm(
    const _Float16* __restrict__ X, const _Float16* __restrict__ W,
    _Float16* __restrict__ Q, _Float16* __restrict__ K,
    _Float16* __restrict__ V) {
  __shared__ _Float16 As[2][64 * 32];
  __shared__ _Float16 Bs[2][128 * 32];
  const int tid = threadIdx.x;
  const int wid = tid >> 5;
  const int lane = tid & 31;
  const int bm0 = blockIdx.y * 64;
  const int bn0 = blockIdx.x * 128;
  const int wm = (wid & 3) * 16;
  const int wn = (wid >> 2) * 64;

  v8f vzero = {0, 0, 0, 0, 0, 0, 0, 0};
  v8f acc[4];
#pragma unroll
  for (int t = 0; t < 4; ++t) acc[t] = vzero;

#if HAVE_TDM
  if (wid == 0) {
    tdm_load_2d(X + (size_t)bm0 * HID, lds_off(As[0]), HID, MROWS - bm0,
                32, 64, HID);
    tdm_load_2d(W + (size_t)bn0 * HID, lds_off(Bs[0]), HID, NQKV - bn0,
                32, 128, HID);
  }
#endif
  int buf = 0;
  for (int k0 = 0; k0 < HID; k0 += 32, buf ^= 1) {
#if HAVE_TDM
    if (wid == 0) __builtin_amdgcn_s_wait_tensorcnt(0);
    __syncthreads();
    if (wid == 0 && k0 + 32 < HID) {   // prefetch next K-slab via TDM
      tdm_load_2d(X + (size_t)bm0 * HID + k0 + 32, lds_off(As[buf ^ 1]),
                  HID - k0 - 32, MROWS - bm0, 32, 64, HID);
      tdm_load_2d(W + (size_t)bn0 * HID + k0 + 32, lds_off(Bs[buf ^ 1]),
                  HID - k0 - 32, NQKV - bn0, 32, 128, HID);
    }
#else
    __syncthreads();
    stage_tile(X + (size_t)bm0 * HID + k0, HID, As[buf], 256, 4);
    stage_tile(W + (size_t)bn0 * HID + k0, HID, Bs[buf], 512, 4);
    __syncthreads();
#endif
    v16h a = load_frag_a(As[buf] + wm * 32, 32);
#pragma unroll
    for (int t = 0; t < 4; ++t) {
      v16h b = load_frag_b_nmajor(Bs[buf] + (wn + t * 16) * 32, 32);
      acc[t] = wmma_f16(a, b, acc[t]);
    }
    __syncthreads();
  }

  const int nloc = lane & 15;
  const int mb8 = (lane >> 4) * 8;
#pragma unroll
  for (int t = 0; t < 4; ++t) {
#pragma unroll
    for (int i = 0; i < 8; ++i) {
      int m = bm0 + wm + mb8 + i;
      int o = bn0 + wn + t * 16 + nloc;
      int b = m >> 11;
      int s = m & (SS - 1);
      int proj = o >> 11;
      int rem = o & (HID - 1);
      int nh = rem >> 7;
      int dh = rem & (DH - 1);
      _Float16* dst = (proj == 0) ? Q : (proj == 1) ? K : V;
      dst[(((size_t)(b * NHEAD + nh)) * SS + s) * DH + dh] = (_Float16)acc[t][i];
    }
  }
}

// ---------------- Kernel 3: RoPE on Q and K (in place, f16) -----------------

__global__ void rope_kernel(_Float16* __restrict__ Q, _Float16* __restrict__ K) {
  int idx = blockIdx.x * blockDim.x + threadIdx.x;
  if (idx >= BB * NHEAD * SS * 64) return;
  int i = idx & 63;
  int s = (idx >> 6) & (SS - 1);
  int bh = idx >> 17;
  size_t base = ((size_t)bh * SS + s) * DH;
  float invf = __expf(-__logf(10000.0f) * (float)i * (1.0f / 64.0f));
  float ang = (float)s * invf;
  float sn, cs;
  __sincosf(ang, &sn, &cs);
  float q0 = (float)Q[base + i], q1 = (float)Q[base + i + 64];
  Q[base + i]      = (_Float16)(q0 * cs - q1 * sn);
  Q[base + i + 64] = (_Float16)(q1 * cs + q0 * sn);
  float k0 = (float)K[base + i], k1 = (float)K[base + i + 64];
  K[base + i]      = (_Float16)(k0 * cs - k1 * sn);
  K[base + i + 64] = (_Float16)(k1 * cs + k0 * sn);
}

// ---------------- Kernel 4: causal flash attention --------------------------
// One block per (b, head, 64-query block). Q frags hoisted to registers;
// K/V key blocks staged to LDS by TDM (double-buffered); online softmax;
// O accumulated in WMMA f32 registers; writes Oh [b,s,H] f16.
__global__ __launch_bounds__(256) void flash_attn(
    const _Float16* __restrict__ Qh, const _Float16* __restrict__ Kh,
    const _Float16* __restrict__ Vh, _Float16* __restrict__ Oh) {
  __shared__ _Float16 Kt[2][64 * DH];   // key rows (row-major, ld=128)
  __shared__ _Float16 Vt[2][64 * DH];   // value rows (row-major, ld=128)
  __shared__ float    sS[64 * 68];      // scores 64x64 (+pad)
  __shared__ _Float16 pS[64 * 72];      // probabilities 64x64 (+pad)
  __shared__ float rowM[64], rowL[64], rowA[64];

  const int tid = threadIdx.x, wid = tid >> 5, lane = tid & 31;
  const int qb = blockIdx.x, nh = blockIdx.y, b = blockIdx.z;
  const int qb0 = qb * 64;
  const size_t hb = ((size_t)(b * NHEAD + nh)) * SS * DH;
  const _Float16* Qp = Qh + hb;
  const _Float16* Kp = Kh + hb;
  const _Float16* Vp = Vh + hb;

  const int s_mt  = wid >> 1;        // m-tile 0..3 (scores and O)
  const int s_nt0 = (wid & 1) * 2;   // score n-tile pair
  const int o_nc0 = (wid & 1) * 64;  // O dh-column half

  v8f vzero = {0, 0, 0, 0, 0, 0, 0, 0};
  v8f oacc[4];
#pragma unroll
  for (int t = 0; t < 4; ++t) oacc[t] = vzero;
  if (tid < 64) { rowM[tid] = -INFINITY; rowL[tid] = 0.0f; }

  // Hoist Q fragments (loop invariant): 4 k-steps over DH.
  v16h qfr[4];
#pragma unroll
  for (int kk = 0; kk < 4; ++kk)
    qfr[kk] = load_frag_a(Qp + (size_t)(qb0 + s_mt * 16) * DH + kk * 32, DH);

  const float scale = 0.08838834764831845f;  // 1/sqrt(128)
  const int nloc = lane & 15;
  const int mb8 = (lane >> 4) * 8;

#if HAVE_TDM
  if (wid == 0) {
    tdm_load_2d(Kp, lds_off(Kt[0]), DH, SS, DH, 64, DH);
    tdm_load_2d(Vp, lds_off(Vt[0]), DH, SS, DH, 64, DH);
  }
#endif
  int buf = 0;
  for (int kb = 0; kb <= qb; ++kb, buf ^= 1) {
    const int kb0 = kb * 64;
#if HAVE_TDM
    if (wid == 0) __builtin_amdgcn_s_wait_tensorcnt(0);
    __syncthreads();
    if (wid == 0 && kb < qb) {       // DMA next key block while computing
      tdm_load_2d(Kp + (size_t)(kb0 + 64) * DH, lds_off(Kt[buf ^ 1]),
                  DH, SS - kb0 - 64, DH, 64, DH);
      tdm_load_2d(Vp + (size_t)(kb0 + 64) * DH, lds_off(Vt[buf ^ 1]),
                  DH, SS - kb0 - 64, DH, 64, DH);
    }
#else
    __syncthreads();
    stage_tile(Kp + (size_t)kb0 * DH, DH, Kt[buf], 1024, 16);
    stage_tile(Vp + (size_t)kb0 * DH, DH, Vt[buf], 1024, 16);
    __syncthreads();
#endif
    // ---- scores: S = scale * Q K^T (each wave: 2 tiles x 4 k-steps) ----
#pragma unroll
    for (int nt = 0; nt < 2; ++nt) {
      v8f sacc = vzero;
      const int n0 = (s_nt0 + nt) * 16;
#pragma unroll
      for (int kk = 0; kk < 4; ++kk) {
        v16h bfr = load_frag_b_nmajor(Kt[buf] + n0 * DH + kk * 32, DH);
        sacc = wmma_f16(qfr[kk], bfr, sacc);
      }
#pragma unroll
      for (int i = 0; i < 8; ++i)
        sS[(s_mt * 16 + mb8 + i) * 68 + n0 + nloc] = sacc[i] * scale;
    }
    __syncthreads();
    // ---- online softmax: one thread per query row ----
    if (tid < 64) {
      const int r = tid;
      const int qg = qb0 + r;
      float mOld = rowM[r];
      float mNew = mOld;
#pragma unroll 8
      for (int c = 0; c < 64; ++c) {
        float v = ((kb0 + c) <= qg) ? sS[r * 68 + c] : -INFINITY;
        mNew = fmaxf(mNew, v);
      }
      float alpha = __expf(mOld - mNew);
      float lsum = 0.0f;
#pragma unroll 8
      for (int c = 0; c < 64; ++c) {
        float v = ((kb0 + c) <= qg) ? sS[r * 68 + c] : -INFINITY;
        float p = __expf(v - mNew);
        lsum += p;
        pS[r * 72 + c] = (_Float16)p;
      }
      rowM[r] = mNew;
      rowL[r] = alpha * rowL[r] + lsum;
      rowA[r] = alpha;
    }
    __syncthreads();
    // ---- rescale O and accumulate O += P @ V ----
#pragma unroll
    for (int t = 0; t < 4; ++t) {
#pragma unroll
      for (int i = 0; i < 8; ++i)
        oacc[t][i] *= rowA[s_mt * 16 + mb8 + i];
    }
#pragma unroll
    for (int kk = 0; kk < 64; kk += 32) {
      v16h a = load_frag_a(pS + s_mt * 16 * 72 + kk, 72);
#pragma unroll
      for (int t = 0; t < 4; ++t) {
        v16h bfr = load_frag_b_kmajor(Vt[buf] + kk * DH + o_nc0 + t * 16, DH);
        oacc[t] = wmma_f16(a, bfr, oacc[t]);
      }
    }
    __syncthreads();
  }

  // ---- normalize and write Oh [b, s, H], h = nh*DH + dh ----
#pragma unroll
  for (int t = 0; t < 4; ++t) {
#pragma unroll
    for (int i = 0; i < 8; ++i) {
      int r = s_mt * 16 + mb8 + i;
      int s = qb0 + r;
      float inv = 1.0f / rowL[r];
      int col = nh * DH + o_nc0 + t * 16 + nloc;
      Oh[((size_t)(b * SS + s)) * HID + col] = (_Float16)(oacc[t][i] * inv);
    }
  }
}

// ---------------- Kernel 5: output projection (NT GEMM, fp32 out) -----------

__global__ __launch_bounds__(256) void oproj_gemm(
    const _Float16* __restrict__ A, const _Float16* __restrict__ W,
    float* __restrict__ Out) {
  __shared__ _Float16 As[2][64 * 32];
  __shared__ _Float16 Bs[2][128 * 32];
  const int tid = threadIdx.x;
  const int wid = tid >> 5;
  const int lane = tid & 31;
  const int bm0 = blockIdx.y * 64;
  const int bn0 = blockIdx.x * 128;
  const int wm = (wid & 3) * 16;
  const int wn = (wid >> 2) * 64;

  v8f vzero = {0, 0, 0, 0, 0, 0, 0, 0};
  v8f acc[4];
#pragma unroll
  for (int t = 0; t < 4; ++t) acc[t] = vzero;

#if HAVE_TDM
  if (wid == 0) {
    tdm_load_2d(A + (size_t)bm0 * HID, lds_off(As[0]), HID, MROWS - bm0,
                32, 64, HID);
    tdm_load_2d(W + (size_t)bn0 * HID, lds_off(Bs[0]), HID, HID - bn0,
                32, 128, HID);
  }
#endif
  int buf = 0;
  for (int k0 = 0; k0 < HID; k0 += 32, buf ^= 1) {
#if HAVE_TDM
    if (wid == 0) __builtin_amdgcn_s_wait_tensorcnt(0);
    __syncthreads();
    if (wid == 0 && k0 + 32 < HID) {
      tdm_load_2d(A + (size_t)bm0 * HID + k0 + 32, lds_off(As[buf ^ 1]),
                  HID - k0 - 32, MROWS - bm0, 32, 64, HID);
      tdm_load_2d(W + (size_t)bn0 * HID + k0 + 32, lds_off(Bs[buf ^ 1]),
                  HID - k0 - 32, HID - bn0, 32, 128, HID);
    }
#else
    __syncthreads();
    stage_tile(A + (size_t)bm0 * HID + k0, HID, As[buf], 256, 4);
    stage_tile(W + (size_t)bn0 * HID + k0, HID, Bs[buf], 512, 4);
    __syncthreads();
#endif
    v16h a = load_frag_a(As[buf] + wm * 32, 32);
#pragma unroll
    for (int t = 0; t < 4; ++t) {
      v16h b = load_frag_b_nmajor(Bs[buf] + (wn + t * 16) * 32, 32);
      acc[t] = wmma_f16(a, b, acc[t]);
    }
    __syncthreads();
  }

  const int nloc = lane & 15;
  const int mb8 = (lane >> 4) * 8;
#pragma unroll
  for (int t = 0; t < 4; ++t) {
#pragma unroll
    for (int i = 0; i < 8; ++i) {
      int m = bm0 + wm + mb8 + i;
      int n = bn0 + wn + t * 16 + nloc;
      Out[(size_t)m * HID + n] = acc[t][i];
    }
  }
}

// ---------------- Host launcher ---------------------------------------------

extern "C" void kernel_launch(void* const* d_in, const int* in_sizes, int n_in,
                              void* d_out, int out_size, void* d_ws,
                              size_t ws_size, hipStream_t stream) {
  (void)in_sizes; (void)n_in; (void)out_size; (void)ws_size;
  const float* x     = (const float*)d_in[0];  // [B,S,H]
  const float* w_qkv = (const float*)d_in[1];  // [3H,H]
  const float* w_o   = (const float*)d_in[2];  // [H,H]
  float* out = (float*)d_out;                  // [B,S,H]

  _Float16* wsh = (_Float16*)d_ws;
  const size_t nX  = (size_t)MROWS * HID;           // 8 Mi
  const size_t nWq = (size_t)NQKV * HID;            // 12 Mi
  const size_t nWo = (size_t)HID * HID;             // 4 Mi
  const size_t nHd = (size_t)BB * NHEAD * SS * DH;  // 8 Mi
  _Float16* Xh  = wsh;
  _Float16* Wqh = Xh + nX;
  _Float16* Woh = Wqh + nWq;
  _Float16* Qh  = Woh + nWo;
  _Float16* Kh  = Qh + nHd;
  _Float16* Vh  = Kh + nHd;
  _Float16* Oh  = Vh + nHd;

  cvt_f32_f16<<<(int)(nX / 4 / 256), 256, 0, stream>>>(x, Xh, (int)(nX / 4));
  cvt_f32_f16<<<(int)(nWq / 4 / 256), 256, 0, stream>>>(w_qkv, Wqh, (int)(nWq / 4));
  cvt_f32_f16<<<(int)(nWo / 4 / 256), 256, 0, stream>>>(w_o, Woh, (int)(nWo / 4));

  qkv_gemm<<<dim3(NQKV / 128, MROWS / 64), 256, 0, stream>>>(Xh, Wqh, Qh, Kh, Vh);
  rope_kernel<<<(BB * NHEAD * SS * 64) / 256, 256, 0, stream>>>(Qh, Kh);
  flash_attn<<<dim3(SS / 64, NHEAD, BB), 256, 0, stream>>>(Qh, Kh, Vh, Oh);
  oproj_gemm<<<dim3(HID / 128, MROWS / 64), 256, 0, stream>>>(Oh, Woh, out);
}
